// CausalSelfAttention_1941325217961
// MI455X (gfx1250) — matrix operand order repaired and checked by reading, not compiled, not run on hardware
//
#include <hip/hip_runtime.h>

// ---------------------------------------------------------------------------
// CDNA5 (gfx1250) causal self-attention, bf16 WMMA pipeline, wave32.
// ---------------------------------------------------------------------------

typedef __attribute__((ext_vector_type(16))) __bf16 v16bf;
typedef __attribute__((ext_vector_type(8)))  __bf16 v8bf;
typedef __attribute__((ext_vector_type(4)))  __bf16 v4bf;
typedef __attribute__((ext_vector_type(8)))  float  v8f;
typedef __attribute__((ext_vector_type(4)))  int    v4i;

#define WMMA_BF16(a, b, c) \
  __builtin_amdgcn_wmma_f32_16x16x32_bf16(false, (a), false, (b), (short)0, (c), false, false)

#ifndef __has_builtin
#define __has_builtin(x) 0
#endif
#if __has_builtin(__builtin_amdgcn_global_load_async_to_lds_b128) && \
    __has_builtin(__builtin_amdgcn_s_wait_asynccnt)
#define HAVE_ASYNC_LDS 1
#endif

union ABFrag { v16bf v; v8bf h[2]; };

// A-matrix (16 x 32, bf16): lane r=lane&15 holds row r.
//   elems 0..7  = K (lane>>4)*8 + 0..7 ; elems 8..15 = K (lane>>4)*8 + 16..23
__device__ __forceinline__ v16bf load_frag_a_g(const __bf16* __restrict__ p,
                                               int ld, int row0, int k0) {
  int lane = threadIdx.x & 31;
  int r = lane & 15, hi = lane >> 4;
  const __bf16* q = p + (size_t)(row0 + r) * ld + k0 + hi * 8;
  ABFrag u;
  u.h[0] = *(const v8bf*)(q);
  u.h[1] = *(const v8bf*)(q + 16);
  return u.v;
}

// B-matrix (32 x 16, bf16), B[k][n], source stored as rows n with k contiguous
// (p[n*ld + k]).  lane n=lane&15 holds column n, K = (lane>>4)*8 + 0..7, 16..23.
__device__ __forceinline__ v16bf load_frag_b_g(const __bf16* __restrict__ p,
                                               int ld, int col0, int k0) {
  int lane = threadIdx.x & 31;
  int n = lane & 15, hi = lane >> 4;
  const __bf16* q = p + (size_t)(col0 + n) * ld + k0 + hi * 8;
  ABFrag u;
  u.h[0] = *(const v8bf*)(q);
  u.h[1] = *(const v8bf*)(q + 16);
  return u.v;
}

// Same as load_frag_b_g but for an LDS tile with arbitrary pitch.
__device__ __forceinline__ v16bf load_frag_b_lds(const __bf16* p, int pitch,
                                                 int col0, int k0) {
  int lane = threadIdx.x & 31;
  int n = lane & 15, hi = lane >> 4;
  const __bf16* q = p + (col0 + n) * pitch + k0 + hi * 8;
  ABFrag u;
  u.h[0] = *(const v8bf*)(q);
  u.h[1] = *(const v8bf*)(q + 16);
  return u.v;
}

// Move 8 bf16 (16 B) global -> LDS.  Uses the CDNA5 async-to-LDS path when the
// toolchain exposes it (tracked by ASYNCcnt), else a plain load + ds_store.
// The builtin takes AS(1)/AS(3) pointers to 128-bit vectors; cast via integers
// (flat LDS address truncates to the 32-bit LDS offset per the aperture rules).
__device__ __forceinline__ void stage16(const __bf16* __restrict__ g, __bf16* l) {
#ifdef HAVE_ASYNC_LDS
  typedef __attribute__((address_space(1))) v4i* gptr_t;
  typedef __attribute__((address_space(3))) v4i* lptr_t;
  __builtin_amdgcn_global_load_async_to_lds_b128(
      (gptr_t)(unsigned long long)g,
      (lptr_t)(unsigned int)(unsigned long long)l,
      0, 0);
#else
  *(v8bf*)l = *(const v8bf*)g;
#endif
}

__device__ __forceinline__ void stage_wait() {
#ifdef HAVE_ASYNC_LDS
  __builtin_amdgcn_s_wait_asynccnt(0);
#endif
  __syncthreads();
}

// ---------------------------------------------------------------------------
// fp32 -> bf16 bulk convert (4 elements / thread)
// ---------------------------------------------------------------------------
__global__ __launch_bounds__(256) void f2bf_kernel(const float* __restrict__ in,
                                                   __bf16* __restrict__ out, int n) {
  int i = (blockIdx.x * 256 + threadIdx.x) * 4;
  if (i < n) {
    float4 v = *(const float4*)(in + i);
    v4bf o;
    o.x = (__bf16)v.x; o.y = (__bf16)v.y; o.z = (__bf16)v.z; o.w = (__bf16)v.w;
    *(v4bf*)(out + i) = o;
  }
}

// ---------------------------------------------------------------------------
// QKV projection: qkv[m,n] = X[m,:].W[n,:] + bias[n]   (M=8192, N=3072, K=1024)
// One wave computes 32 x 64 (8 accumulators; 2 A frags share each B frag).
// Q section pre-scaled by 1/sqrt(D)=0.125 (bias included, matching reference).
// ---------------------------------------------------------------------------
__global__ __launch_bounds__(256) void qkv_gemm(const __bf16* __restrict__ X,
                                                const __bf16* __restrict__ W,
                                                const float* __restrict__ bias,
                                                __bf16* __restrict__ out) {
  const int N = 3072, K = 1024;
  int wid = blockIdx.x * 8 + (threadIdx.x >> 5);
  int mT = wid / (N / 64), nG = wid % (N / 64);
  int m0 = mT * 32, n0 = nG * 64;
  int lane = threadIdx.x & 31, hi = lane >> 4, nl = lane & 15;

  v8f acc[2][4] = {};
  for (int k0 = 0; k0 < K; k0 += 32) {
    v16bf a0 = load_frag_a_g(X, K, m0, k0);
    v16bf a1 = load_frag_a_g(X, K, m0 + 16, k0);
#pragma unroll
    for (int j = 0; j < 4; ++j) {
      v16bf b = load_frag_b_g(W, K, n0 + j * 16, k0);
      acc[0][j] = WMMA_BF16(a0, b, acc[0][j]);
      acc[1][j] = WMMA_BF16(a1, b, acc[1][j]);
    }
  }
  float scl = (n0 < 1024) ? 0.125f : 1.0f;
#pragma unroll
  for (int j = 0; j < 4; ++j) {
    int n = n0 + j * 16 + nl;
    float bv = bias[n];
#pragma unroll
    for (int r = 0; r < 2; ++r)
#pragma unroll
      for (int i = 0; i < 8; ++i)
        out[(size_t)(m0 + r * 16 + i + hi * 8) * N + n] = (__bf16)((acc[r][j][i] + bv) * scl);
  }
}

// ---------------------------------------------------------------------------
// V transpose: qkv V section [b,t, 2048+h*64+d] -> Vt[((b*16+h)*64+d)*2048 + t]
// ---------------------------------------------------------------------------
__global__ __launch_bounds__(256) void v_transpose(const __bf16* __restrict__ qkv,
                                                   __bf16* __restrict__ vt) {
  __shared__ __bf16 tile[64][66];
  int blk = blockIdx.x;                 // B*H*(T/64) = 2048
  int tb = blk & 31, bh = blk >> 5;
  int b = bh >> 4, h = bh & 15;
  int t0 = tb * 64;
  const __bf16* src = qkv + ((size_t)b * 2048 + t0) * 3072 + 2048 + h * 64;
  for (int idx = threadIdx.x; idx < 64 * 64; idx += 256) {
    int t = idx >> 6, d = idx & 63;
    tile[t][d] = src[(size_t)t * 3072 + d];
  }
  __syncthreads();
  __bf16* dst = vt + ((size_t)(b * 16 + h) * 64) * 2048 + t0;
  for (int idx = threadIdx.x; idx < 64 * 64; idx += 256) {
    int d = idx >> 6, t = idx & 63;
    dst[(size_t)d * 2048 + t] = tile[t][d];
  }
}

// ---------------------------------------------------------------------------
// Flash attention.  Block = 8 waves = 128 queries of one (b,h).
// Block-uniform key loop: waves cooperatively stage K[64x64] and Vt[64x64]
// tiles into double-buffered LDS (async-to-LDS when available), then each
// wave computes its 16-query tile from LDS fragments.  Waves past their
// causal extent skip the math but keep hitting the barrier.
// ---------------------------------------------------------------------------
__global__ __launch_bounds__(256) void attn_kernel(const __bf16* __restrict__ qkv,
                                                   const __bf16* __restrict__ vt,
                                                   const int* __restrict__ amask,
                                                   __bf16* __restrict__ yb) {
  const int T = 2048, CQ = 3072, PITCH = 72;
  __shared__ __bf16 kbuf[2][64][PITCH];   // [key][d]
  __shared__ __bf16 vbuf[2][64][PITCH];   // [d][key]
  __shared__ __bf16 pbuf[8][16 * 64];     // per-wave P staging (C -> A layout)

  int w = threadIdx.x >> 5;
  int lane = threadIdx.x & 31, hi = lane >> 4, nl = lane & 15;
  int blk = blockIdx.x;                  // 64 (b,h) * 16 query blocks = 1024
  int qB = blk & 15, bh = blk >> 4;
  int b = bh >> 4, h = bh & 15;
  int q0B = qB * 128;
  int q0 = q0B + w * 16;

  const __bf16* qrow = qkv + (size_t)b * T * CQ;               // Q at col h*64
  const __bf16* ksrc = qkv + (size_t)b * T * CQ + 1024 + h * 64;
  const __bf16* vsrc = vt + (size_t)bh * 64 * T;               // rows d, len T
  const int* mk = amask + b * T;

  // staging mapping: 256 threads, each moves 2x16B for K and for V
  int sr = threadIdx.x >> 2;             // 0..63 (row)
  int sc = (threadIdx.x & 3) * 16;       // 0/16/32/48 (col)

  // Q fragments (A layout), register resident
  v16bf qf[2];
#pragma unroll
  for (int c = 0; c < 2; ++c)
    qf[c] = load_frag_a_g(qrow, CQ, q0, h * 64 + c * 32);

  v8f o[4] = {};
  float mrow[8], lrow[8];
#pragma unroll
  for (int i = 0; i < 8; ++i) { mrow[i] = -1e30f; lrow[i] = 0.f; }

  int nIter = q0B / 64 + 2;              // covers keys up to q0B+127
  for (int it = 0; it < nIter; ++it) {
    int kb = it * 64;
    int buf = it & 1;
    // ---- cooperative staging (double buffered, one barrier / iter) ----
    stage16(ksrc + (size_t)(kb + sr) * CQ + sc,     &kbuf[buf][sr][sc]);
    stage16(ksrc + (size_t)(kb + sr) * CQ + sc + 8, &kbuf[buf][sr][sc + 8]);
    stage16(vsrc + (size_t)sr * T + kb + sc,        &vbuf[buf][sr][sc]);
    stage16(vsrc + (size_t)sr * T + kb + sc + 8,    &vbuf[buf][sr][sc + 8]);
    stage_wait();

    if (kb < q0 + 16) {                  // wave-uniform causal skip
      const __bf16* kt = &kbuf[buf][0][0];
      const __bf16* vtile = &vbuf[buf][0][0];
      // ---- scores S[16q, 64k] ----
      v8f s[4];
#pragma unroll
      for (int j = 0; j < 4; ++j) {
        v8f a = {};
#pragma unroll
        for (int c = 0; c < 2; ++c) {
          v16bf kf = load_frag_b_lds(kt, PITCH, j * 16, c * 32);
          a = WMMA_BF16(qf[c], kf, a);
        }
        s[j] = a;
      }
      // ---- causal + key-padding mask ----
#pragma unroll
      for (int j = 0; j < 4; ++j) {
        int kg = kb + j * 16 + nl;
        bool pad_ok = (mk[kg] != 0);
#pragma unroll
        for (int i = 0; i < 8; ++i) {
          int qg = q0 + i + hi * 8;
          if (!(pad_ok && kg <= qg)) s[j][i] = -1e30f;
        }
      }
      // ---- online softmax ----
      float mnew[8], alpha[8], rs[8];
#pragma unroll
      for (int i = 0; i < 8; ++i) {
        float v = fmaxf(fmaxf(s[0][i], s[1][i]), fmaxf(s[2][i], s[3][i]));
        v = fmaxf(v, __shfl_xor(v, 1));
        v = fmaxf(v, __shfl_xor(v, 2));
        v = fmaxf(v, __shfl_xor(v, 4));
        v = fmaxf(v, __shfl_xor(v, 8));
        mnew[i] = fmaxf(mrow[i], v);
        alpha[i] = __expf(mrow[i] - mnew[i]);
        rs[i] = 0.f;
      }
#pragma unroll
      for (int j = 0; j < 4; ++j)
#pragma unroll
        for (int i = 0; i < 8; ++i) {
          float p = __expf(s[j][i] - mnew[i]);
          s[j][i] = p;
          rs[i] += p;
        }
#pragma unroll
      for (int i = 0; i < 8; ++i) {
        float v = rs[i];
        v += __shfl_xor(v, 1);
        v += __shfl_xor(v, 2);
        v += __shfl_xor(v, 4);
        v += __shfl_xor(v, 8);
        lrow[i] = lrow[i] * alpha[i] + v;
        mrow[i] = mnew[i];
      }
#pragma unroll
      for (int jd = 0; jd < 4; ++jd)
#pragma unroll
        for (int i = 0; i < 8; ++i)
          o[jd][i] *= alpha[i];

      // ---- P: C-layout regs -> wave-private LDS (row-major 16x64) ----
      __bf16* pl = pbuf[w];
#pragma unroll
      for (int j = 0; j < 4; ++j)
#pragma unroll
        for (int i = 0; i < 8; ++i)
          pl[(i + hi * 8) * 64 + j * 16 + nl] = (__bf16)s[j][i];

      // ---- O += P @ V ----
#pragma unroll
      for (int c = 0; c < 2; ++c) {
        const __bf16* ap = pl + (size_t)nl * 64 + c * 32 + hi * 8;
        ABFrag u;
        u.h[0] = *(const v8bf*)(ap);
        u.h[1] = *(const v8bf*)(ap + 16);
        v16bf pf = u.v;
#pragma unroll
        for (int jd = 0; jd < 4; ++jd) {
          v16bf vf = load_frag_b_lds(vtile, PITCH, jd * 16, c * 32);
          o[jd] = WMMA_BF16(pf, vf, o[jd]);
        }
      }
    }
  }

  // ---- normalize and store Y[b,t, h*64+d] (bf16) ----
  __bf16* ybase = yb + ((size_t)b * T + q0) * 1024 + h * 64;
#pragma unroll
  for (int i = 0; i < 8; ++i) {
    float inv = 1.0f / lrow[i];
#pragma unroll
    for (int jd = 0; jd < 4; ++jd)
      ybase[(size_t)(i + hi * 8) * 1024 + jd * 16 + nl] = (__bf16)(o[jd][i] * inv);
  }
}

// ---------------------------------------------------------------------------
// Output projection: out[m,n] = Y[m,:].Wout[n,:]  (M=8192, N=K=1024, fp32 out)
// ---------------------------------------------------------------------------
__global__ __launch_bounds__(256) void out_gemm(const __bf16* __restrict__ Y,
                                                const __bf16* __restrict__ W,
                                                float* __restrict__ out) {
  const int N = 1024, K = 1024;
  int wid = blockIdx.x * 8 + (threadIdx.x >> 5);
  int mT = wid / (N / 64), nG = wid % (N / 64);
  int m0 = mT * 32, n0 = nG * 64;
  int lane = threadIdx.x & 31, hi = lane >> 4, nl = lane & 15;

  v8f acc[2][4] = {};
  for (int k0 = 0; k0 < K; k0 += 32) {
    v16bf a0 = load_frag_a_g(Y, K, m0, k0);
    v16bf a1 = load_frag_a_g(Y, K, m0 + 16, k0);
#pragma unroll
    for (int j = 0; j < 4; ++j) {
      v16bf b = load_frag_b_g(W, K, n0 + j * 16, k0);
      acc[0][j] = WMMA_BF16(a0, b, acc[0][j]);
      acc[1][j] = WMMA_BF16(a1, b, acc[1][j]);
    }
  }
#pragma unroll
  for (int j = 0; j < 4; ++j) {
    int n = n0 + j * 16 + nl;
#pragma unroll
    for (int r = 0; r < 2; ++r)
#pragma unroll
      for (int i = 0; i < 8; ++i)
        out[(size_t)(m0 + r * 16 + i + hi * 8) * N + n] = acc[r][j][i];
  }
}

// ---------------------------------------------------------------------------
// Host-side launcher
// ---------------------------------------------------------------------------
extern "C" void kernel_launch(void* const* d_in, const int* in_sizes, int n_in,
                              void* d_out, int out_size, void* d_ws, size_t ws_size,
                              hipStream_t stream) {
  (void)in_sizes; (void)n_in; (void)out_size; (void)ws_size;

  const float* x     = (const float*)d_in[0];   // [4,2048,1024]
  const int*   amask = (const int*)d_in[1];     // [4,2048]
  const float* Wqkv  = (const float*)d_in[2];   // [3072,1024]
  const float* bqkv  = (const float*)d_in[3];   // [3072]
  const float* Wout  = (const float*)d_in[4];   // [1024,1024]
  float*       out   = (float*)d_out;           // [4,2048,1024]

  const int nX  = 4 * 2048 * 1024;   // 8388608
  const int nWq = 3072 * 1024;       // 3145728
  const int nWo = 1024 * 1024;       // 1048576

  char* ws = (char*)d_ws;
  __bf16* xb    = (__bf16*)(ws);               // 16 MB (reused as yb)
  __bf16* wqkvb = (__bf16*)(ws + 16777216);    //  6 MB
  __bf16* woutb = (__bf16*)(ws + 23068672);    //  2 MB
  __bf16* qkv   = (__bf16*)(ws + 25165824);    // 48 MB  [B,T,3C]
  __bf16* vt    = (__bf16*)(ws + 75497472);    // 16 MB  [B,H,D,T]
  __bf16* yb    = xb;

  f2bf_kernel<<<nX  / 1024, 256, 0, stream>>>(x,    xb,    nX);
  f2bf_kernel<<<nWq / 1024, 256, 0, stream>>>(Wqkv, wqkvb, nWq);
  f2bf_kernel<<<nWo / 1024, 256, 0, stream>>>(Wout, woutb, nWo);

  qkv_gemm<<<1536, 256, 0, stream>>>(xb, wqkvb, bqkv, qkv);   // 256 x 48 wave tiles
  v_transpose<<<2048, 256, 0, stream>>>(qkv, vt);
  attn_kernel<<<1024, 256, 0, stream>>>(qkv, vt, amask, yb);
  out_gemm<<<512, 256, 0, stream>>>(yb, woutb, out);          // 256 x 16 wave tiles
}
